// LowRankAttention_63513976373303
// MI455X (gfx1250) — matrix operand order repaired and checked by reading, not compiled, not run on hardware
//
#include <hip/hip_runtime.h>
#include <hip/hip_bf16.h>

typedef _Float16 f16;
typedef __attribute__((ext_vector_type(16))) _Float16 v16h;
typedef __attribute__((ext_vector_type(8)))  _Float16 v8h;
typedef __attribute__((ext_vector_type(8)))  float    v8f;

union Frag16 {
    v16h v;
    v8h  h[2];
    _Float16 e[16];
};

static __device__ __forceinline__ v8f wmma_f16(v16h a, v16h b, v8f c) {
    // D = A(16x32 f16) * B(32x16 f16) + C(16x16 f32)
    return __builtin_amdgcn_wmma_f32_16x16x32_f16(
        /*neg_a=*/false, a, /*neg_b=*/false, b,
        /*c_mod=*/(short)0, c, /*reuse_a=*/false, /*reuse_b=*/false);
}

#define LOAD_FRAG32B(frag, ptr)                      \
    do {                                             \
        (frag).h[0] = ((const v8h*)(ptr))[0];        \
        (frag).h[1] = ((const v8h*)(ptr))[1];        \
    } while (0)

// ---------------------------------------------------------------------------
// fp32 -> f16 elementwise convert (8 elements / thread)
// ---------------------------------------------------------------------------
__global__ void lra_cvt_f16(const float* __restrict__ in, f16* __restrict__ out, int n8) {
    int t = blockIdx.x * blockDim.x + threadIdx.x;
    if (t >= n8) return;
    const float4* p = (const float4*)(in + (size_t)t * 8);
    float4 a = p[0], b = p[1];
    v8h o;
    o[0] = (f16)a.x; o[1] = (f16)a.y; o[2] = (f16)a.z; o[3] = (f16)a.w;
    o[4] = (f16)b.x; o[5] = (f16)b.y; o[6] = (f16)b.z; o[7] = (f16)b.w;
    *(v8h*)(out + (size_t)t * 8) = o;
}

// ---------------------------------------------------------------------------
// Pack fp32 weight W[K,N] (row major) into WMMA B-fragment order, f16:
//   Wp[kt][nt][lane][e] = W[kt*32 + (lane/16)*8 + e + (e&8)][nt*16 + lane%16]
// (table order: lanes 0-15 hold K=0..7,16..23, lanes 16-31 K=8..15,24..31)
// ---------------------------------------------------------------------------
__global__ void lra_pack_w(const float* __restrict__ W, f16* __restrict__ Wp,
                           int N, int K) {
    int total = (K >> 5) * (N >> 4) * 32;
    int tid = blockIdx.x * blockDim.x + threadIdx.x;
    if (tid >= total) return;
    int lane = tid & 31;
    int g = (lane >> 4) & 1;
    int nc = lane & 15;
    int rest = tid >> 5;
    int NT = N >> 4;
    int nt = rest % NT;
    int kt = rest / NT;
    const float* src = W + (size_t)(kt * 32 + g * 8) * N + nt * 16 + nc;
    Frag16 o;
#pragma unroll
    for (int e = 0; e < 16; ++e) o.e[e] = (f16)src[(size_t)(e + (e & 8)) * N];
    v8h* dst = (v8h*)(Wp + (size_t)tid * 16);
    dst[0] = o.h[0];
    dst[1] = o.h[1];
}

// ---------------------------------------------------------------------------
// Pack f16 V[tok, E] (row major, col = h*64+d) into attention B-fragment order:
//   Vp[((b*H+h)*JT + jt)*4 + dsub][lane][e] =
//       V[b, j = jt*32 + (lane/16)*8 + e + (e&8), h, dsub*16 + lane%16]
// ---------------------------------------------------------------------------
__global__ void lra_pack_v(const f16* __restrict__ V, f16* __restrict__ Vp,
                           int B_, int S, int H) {
    int JT = S >> 5;
    int total = B_ * H * JT * 4 * 32;
    int tid = blockIdx.x * blockDim.x + threadIdx.x;
    if (tid >= total) return;
    int lane = tid & 31;
    int g = (lane >> 4) & 1;
    int dc = lane & 15;
    int rest = tid >> 5;
    int dsub = rest & 3; rest >>= 2;
    int jt = rest % JT; rest /= JT;
    int h = rest % H;
    int b = rest / H;
    const f16* src = V + ((size_t)(b * S + jt * 32 + g * 8)) * (H * 64)
                       + h * 64 + dsub * 16 + dc;
    Frag16 o;
#pragma unroll
    for (int e = 0; e < 16; ++e) o.e[e] = src[(size_t)(e + (e & 8)) * (H * 64)];
    v8h* dst = (v8h*)(Vp + (size_t)tid * 16);
    dst[0] = o.h[0];
    dst[1] = o.h[1];
}

// ---------------------------------------------------------------------------
// GEMM: C[M,N] = A[M,K]f16(row major) x Bp(packed f16) + bias(f32)
// Wave computes 32(M) x 64(N). Fully register double-buffered over K with
// NAMED fragments only (no arrays -> no scratch). N, K compile-time.
// Block = 4 waves -> 128x64 tile (B fragments reused 4x via L1).
// ---------------------------------------------------------------------------
template <typename OutT, int N, int K>
__global__ void lra_gemm(const f16* __restrict__ A, const f16* __restrict__ Bp,
                         const float* __restrict__ bias, OutT* __restrict__ C) {
    constexpr int KT = K / 32;
    constexpr int NT = N / 16;
    constexpr int BKT = NT * 32 * 16;  // halves per kt of packed B
    const int lane = threadIdx.x & 31;
    const int wave = threadIdx.x >> 5;
    const int g = lane >> 4;
    const int ic = lane & 15;
    const int mBase = (blockIdx.y * 4 + wave) * 32;
    const int nBase = blockIdx.x * 64;

    v8f c00{}, c01{}, c02{}, c03{}, c10{}, c11{}, c12{}, c13{};

    // Single base pointer per stream; all sub-tile offsets are constants.
    const f16* aPtr = A + (size_t)(mBase + ic) * K + g * 8;
    const f16* bPtr = Bp + ((size_t)(nBase >> 4) * 32 + lane) * 16;

    Frag16 aA0, aA1, bA0, bA1, bA2, bA3;  // buffer set A
    Frag16 aB0, aB1, bB0, bB1, bB2, bB3;  // buffer set B

#define G_LOADK(kt, a0, a1, b0, b1, b2, b3)                        \
    do {                                                           \
        const f16* pa_ = aPtr + (size_t)(kt) * 32;                 \
        (a0).h[0] = ((const v8h*)pa_)[0];                          \
        (a0).h[1] = *(const v8h*)(pa_ + 16);                       \
        (a1).h[0] = *(const v8h*)(pa_ + 16 * K);                   \
        (a1).h[1] = *(const v8h*)(pa_ + 16 * K + 16);              \
        const f16* pb_ = bPtr + (size_t)(kt) * BKT;                \
        LOAD_FRAG32B(b0, pb_);                                     \
        LOAD_FRAG32B(b1, pb_ + 512);                               \
        LOAD_FRAG32B(b2, pb_ + 1024);                              \
        LOAD_FRAG32B(b3, pb_ + 1536);                              \
    } while (0)

#define G_STEP(kt, aC0, aC1, bC0, bC1, bC2, bC3, aN0, aN1, bN0, bN1, bN2, bN3) \
    do {                                                                       \
        if ((kt) + 1 < KT) G_LOADK((kt) + 1, aN0, aN1, bN0, bN1, bN2, bN3);    \
        if ((kt) + 4 < KT) {                                                   \
            __builtin_prefetch(aPtr + (size_t)((kt) + 4) * 32, 0, 3);          \
            __builtin_prefetch(bPtr + (size_t)((kt) + 4) * BKT, 0, 3);         \
        }                                                                      \
        c00 = wmma_f16((aC0).v, (bC0).v, c00);                                 \
        c10 = wmma_f16((aC1).v, (bC0).v, c10);                                 \
        c01 = wmma_f16((aC0).v, (bC1).v, c01);                                 \
        c11 = wmma_f16((aC1).v, (bC1).v, c11);                                 \
        c02 = wmma_f16((aC0).v, (bC2).v, c02);                                 \
        c12 = wmma_f16((aC1).v, (bC2).v, c12);                                 \
        c03 = wmma_f16((aC0).v, (bC3).v, c03);                                 \
        c13 = wmma_f16((aC1).v, (bC3).v, c13);                                 \
    } while (0)

    G_LOADK(0, aA0, aA1, bA0, bA1, bA2, bA3);
    for (int kt = 0; kt < KT; kt += 2) {
        G_STEP(kt, aA0, aA1, bA0, bA1, bA2, bA3, aB0, aB1, bB0, bB1, bB2, bB3);
        G_STEP(kt + 1, aB0, aB1, bB0, bB1, bB2, bB3, aA0, aA1, bA0, bA1, bA2, bA3);
    }
#undef G_STEP
#undef G_LOADK

    // f32 C/D layout: row = g*8 + r (+16 per m-subtile), col = lane%16
    const float bv0 = bias[nBase + ic];
    const float bv1 = bias[nBase + 16 + ic];
    const float bv2 = bias[nBase + 32 + ic];
    const float bv3 = bias[nBase + 48 + ic];
    OutT* crow = C + (size_t)(mBase + g * 8) * N + nBase + ic;
#pragma unroll
    for (int r = 0; r < 8; ++r) {
        OutT* p0 = crow + (size_t)r * N;
        OutT* p1 = crow + (size_t)(16 + r) * N;
        p0[0]  = (OutT)(c00[r] + bv0);
        p0[16] = (OutT)(c01[r] + bv1);
        p0[32] = (OutT)(c02[r] + bv2);
        p0[48] = (OutT)(c03[r] + bv3);
        p1[0]  = (OutT)(c10[r] + bv0);
        p1[16] = (OutT)(c11[r] + bv1);
        p1[32] = (OutT)(c12[r] + bv2);
        p1[48] = (OutT)(c13[r] + bv3);
    }
}

// ---------------------------------------------------------------------------
// Flash attention per (b, h, 16-query tile). S=2048, H=16, RANK=32, HEAD_DIM=64.
// Scores computed transposed: S^T = K_tile(16x32) x Q^T(32x16) so the P
// fragment for P@V is a same-lane repack of the two score D-fragments.
// K fragments are single-buffered: reloaded (for jt+1) immediately after the
// score WMMAs consume them; the softmax VALU chain + P@V covers the latency.
// ---------------------------------------------------------------------------
__global__ void lra_attn(const f16* __restrict__ Qh, const f16* __restrict__ Kh,
                         const f16* __restrict__ Vp, f16* __restrict__ Oh) {
    constexpr int S = 2048, H = 16;
    constexpr int JT = S / 32;      // 64
    constexpr int LD = H * 32;      // 512 halves per token row of Q/K
    const int lane = threadIdx.x & 31;
    const int wave = threadIdx.x >> 5;
    const int g = lane >> 4;
    const int ic = lane & 15;
    const int it = blockIdx.x * 4 + wave;  // query tile (16 queries)
    const int h = blockIdx.y;
    const int b = blockIdx.z;
    const float scale = 0.17677669529663687f;  // 1/sqrt(32)

    // Q^T B-fragment (col n = query i = lane%16; K-dim r = g*8 + e + (e&8)), with the
    // softmax scale folded in once.
    const f16* qrow = Qh + (size_t)(b * S + it * 16 + ic) * LD + h * 32 + g * 8;
    Frag16 qf;
    qf.h[0] = ((const v8h*)qrow)[0];
    qf.h[1] = *(const v8h*)(qrow + 16);
#pragma unroll
    for (int e = 0; e < 16; ++e) qf.e[e] = (f16)((float)qf.e[e] * scale);

    v8f o0{}, o1{}, o2{}, o3{};
    float m_i = -1e30f, l_i = 0.0f;

    const f16* kPtr = Kh + (size_t)(b * S + ic) * LD + h * 32 + g * 8;
    const f16* vPtr = Vp + ((size_t)(b * H + h) * JT * 4 * 32 + lane) * 16;

    Frag16 kf0, kf1;  // single-buffered K fragments

    // preload K for jt = 0
    kf0.h[0] = ((const v8h*)kPtr)[0];
    kf0.h[1] = *(const v8h*)(kPtr + 16);
    kf1.h[0] = *(const v8h*)(kPtr + 16 * LD);
    kf1.h[1] = *(const v8h*)(kPtr + 16 * LD + 16);

    for (int jt = 0; jt < JT; ++jt) {
        // V loads for this jt: consumed after the softmax chain
        const f16* vb = vPtr + (size_t)jt * 2048;
        Frag16 v0, v1, v2, v3;
        LOAD_FRAG32B(v0, vb);
        LOAD_FRAG32B(v1, vb + 512);
        LOAD_FRAG32B(v2, vb + 1024);
        LOAD_FRAG32B(v3, vb + 1536);

        v8f s0{}, s1{};
        s0 = wmma_f16(kf0.v, qf.v, s0);  // S^T[j,i], j in [0,16)
        s1 = wmma_f16(kf1.v, qf.v, s1);  // S^T[j,i], j in [16,32)

        // K fragments are dead after the score WMMAs read them: start loading
        // jt+1 into the same registers (covered by softmax + P@V below).
        if (jt + 1 < JT) {
            const f16* kr = kPtr + (size_t)(jt + 1) * (32 * LD);
            kf0.h[0] = ((const v8h*)kr)[0];
            kf0.h[1] = *(const v8h*)(kr + 16);
            kf1.h[0] = *(const v8h*)(kr + 16 * LD);
            kf1.h[1] = *(const v8h*)(kr + 16 * LD + 16);
        }

        // lane holds col i = ic; rows j = g*8+r (s0), 16+g*8+r (s1)
        float p0[8], p1[8];
        float mx = -1e30f;
#pragma unroll
        for (int r = 0; r < 8; ++r) {
            p0[r] = s0[r];
            p1[r] = s1[r];
            mx = fmaxf(mx, fmaxf(p0[r], p1[r]));
        }
        mx = fmaxf(mx, __shfl_xor(mx, 16));
        float m_new = fmaxf(m_i, mx);
        float alpha = __expf(m_i - m_new);
        float lsum = 0.0f;
#pragma unroll
        for (int r = 0; r < 8; ++r) {
            p0[r] = __expf(p0[r] - m_new);
            p1[r] = __expf(p1[r] - m_new);
            lsum += p0[r] + p1[r];
        }
        lsum += __shfl_xor(lsum, 16);
        l_i = l_i * alpha + lsum;
        m_i = m_new;

        // P A-fragment: same-lane repack, e<8 <- p0[e], e>=8 <- p1[e-8]
        Frag16 pf;
#pragma unroll
        for (int r = 0; r < 8; ++r) {
            pf.e[r] = (f16)p0[r];
            pf.e[8 + r] = (f16)p1[r];
        }
        // Rescale O accumulators; O rows are i = g*8 + r, alpha lives at lane i
#pragma unroll
        for (int r = 0; r < 8; ++r) {
            float a = __shfl(alpha, g * 8 + r);
            o0[r] *= a; o1[r] *= a; o2[r] *= a; o3[r] *= a;
        }
        o0 = wmma_f16(pf.v, v0.v, o0);
        o1 = wmma_f16(pf.v, v1.v, o1);
        o2 = wmma_f16(pf.v, v2.v, o2);
        o3 = wmma_f16(pf.v, v3.v, o3);
    }

    float linv = 1.0f / l_i;
#pragma unroll
    for (int r = 0; r < 8; ++r) {
        float inv = __shfl(linv, g * 8 + r);
        int row = b * S + it * 16 + g * 8 + r;
        f16* op = Oh + (size_t)row * (H * 64) + h * 64 + ic;
        op[0]  = (f16)(o0[r] * inv);
        op[16] = (f16)(o1[r] * inv);
        op[32] = (f16)(o2[r] * inv);
        op[48] = (f16)(o3[r] * inv);
    }
}

// ---------------------------------------------------------------------------
extern "C" void kernel_launch(void* const* d_in, const int* in_sizes, int n_in,
                              void* d_out, int out_size, void* d_ws, size_t ws_size,
                              hipStream_t stream) {
    (void)n_in; (void)out_size; (void)ws_size;
    const int S = 2048, E = 1024, H = 16;
    const int B_ = in_sizes[0] / (S * E);
    const size_t nTok = (size_t)B_ * S;

    const float* hidden = (const float*)d_in[0];
    const float* Wq = (const float*)d_in[1];
    const float* bq = (const float*)d_in[2];
    const float* Wk = (const float*)d_in[3];
    const float* bk = (const float*)d_in[4];
    const float* Wv = (const float*)d_in[5];
    const float* bv = (const float*)d_in[6];
    const float* Wo = (const float*)d_in[7];
    const float* bo = (const float*)d_in[8];
    float* out = (float*)d_out;

    // Workspace carve-up (f16 elements, all 16B aligned)
    f16* p = (f16*)d_ws;
    f16* Xh  = p; p += nTok * E;        // hidden f16
    f16* Qh  = p; p += nTok * 512;      // q  [tok, H*32]
    f16* Kh  = p; p += nTok * 512;      // k  [tok, H*32]
    f16* Vh  = p; p += nTok * E;        // v  [tok, H*64]
    f16* Vp  = p; p += nTok * E;        // v packed for attention B-fragments
    f16* Ohd = p; p += nTok * E;        // attention output f16
    f16* Wqp = p; p += (size_t)E * 512;
    f16* Wkp = p; p += (size_t)E * 512;
    f16* Wvp = p; p += (size_t)E * E;
    f16* Wop = p; p += (size_t)E * E;

    // 1) hidden -> f16
    {
        int n8 = (int)(nTok * E / 8);
        lra_cvt_f16<<<(n8 + 255) / 256, 256, 0, stream>>>(hidden, Xh, n8);
    }
    // 2) pack weights into B-fragment order
    {
        int t512 = (E / 32) * (512 / 16) * 32;
        int t1024 = (E / 32) * (E / 16) * 32;
        lra_pack_w<<<(t512 + 255) / 256, 256, 0, stream>>>(Wq, Wqp, 512, E);
        lra_pack_w<<<(t512 + 255) / 256, 256, 0, stream>>>(Wk, Wkp, 512, E);
        lra_pack_w<<<(t1024 + 255) / 256, 256, 0, stream>>>(Wv, Wvp, E, E);
        lra_pack_w<<<(t1024 + 255) / 256, 256, 0, stream>>>(Wo, Wop, E, E);
    }
    // 3) projections (block covers 128 rows x 64 cols)
    lra_gemm<f16, 512, 1024><<<dim3(512 / 64, (unsigned)(nTok / 128)), 128, 0, stream>>>(
        Xh, Wqp, bq, Qh);
    lra_gemm<f16, 512, 1024><<<dim3(512 / 64, (unsigned)(nTok / 128)), 128, 0, stream>>>(
        Xh, Wkp, bk, Kh);
    lra_gemm<f16, 1024, 1024><<<dim3(E / 64, (unsigned)(nTok / 128)), 128, 0, stream>>>(
        Xh, Wvp, bv, Vh);
    // 4) pack V for attention
    {
        int tv = B_ * H * (S / 32) * 4 * 32;
        lra_pack_v<<<(tv + 255) / 256, 256, 0, stream>>>(Vh, Vp, B_, S, H);
    }
    // 5) flash attention (wave per (b,h,16-query tile); 4 waves/block)
    lra_attn<<<dim3(S / 16 / 4, H, B_), 128, 0, stream>>>(Qh, Kh, Vp, Ohd);
    // 6) output projection -> f32 + bias
    lra_gemm<float, 1024, 1024><<<dim3(E / 64, (unsigned)(nTok / 128)), 128, 0, stream>>>(
        Ohd, Wop, bo, out);
}
